// LMDGNN_63582695850402
// MI455X (gfx1250) — matrix operand order repaired and checked by reading, not compiled, not run on hardware
//
#include <hip/hip_runtime.h>
#include <hip/hip_bf16.h>
#include <stdint.h>

typedef __bf16 bf16_t;
typedef __attribute__((ext_vector_type(16))) __bf16 v16bf;
typedef __attribute__((ext_vector_type(8)))  float  v8f;

#define LDS_STRIDE 40  // bf16 elems per LDS tile row (80B: keeps 16B chunk alignment, spreads banks)

// ---------------------------------------------------------------------------
// Tiled transpose: dst(N x K) bf16 <- src(K x N) fp32, coalesced both sides
// ---------------------------------------------------------------------------
__global__ __launch_bounds__(256) void transpose_to_bf16(
    const float* __restrict__ src, bf16_t* __restrict__ dst, int K, int N) {
  __shared__ float tile[32][33];
  const int kb = blockIdx.x * 32, nb = blockIdx.y * 32;
  const int tx = threadIdx.x & 31, ty = threadIdx.x >> 5;  // 32 x 8
  #pragma unroll
  for (int yy = ty; yy < 32; yy += 8) {
    const int k = kb + yy, n = nb + tx;
    tile[yy][tx] = (k < K && n < N) ? src[(size_t)k * N + n] : 0.f;
  }
  __syncthreads();
  #pragma unroll
  for (int yy = ty; yy < 32; yy += 8) {
    const int n = nb + yy, k = kb + tx;
    if (n < N && k < K) dst[(size_t)n * K + k] = (bf16_t)tile[tx][yy];
  }
}

// x (rows x srcStride) fp32 -> dst (rows x cols) bf16 (drops the last column)
__global__ void convert_x_bf16(const float* __restrict__ src, bf16_t* __restrict__ dst,
                               int rows, int cols, int srcStride) {
  const int idx = blockIdx.x * 256 + threadIdx.x;
  if (idx >= rows * cols) return;
  const int r = idx / cols, c = idx - r * cols;
  dst[idx] = (bf16_t)src[(size_t)r * srcStride + c];
}

__global__ void gate_bias_kernel(const float* __restrict__ iwb, const float* __restrict__ iub,
                                 const float* __restrict__ owb, const float* __restrict__ oub,
                                 const float* __restrict__ cwb, const float* __restrict__ cub,
                                 float* __restrict__ out) {
  const int t = threadIdx.x;            // 768 threads
  const int g = t >> 8, j = t & 255;
  float v;
  if (g == 0)      v = iwb[j] + iub[j];
  else if (g == 1) v = owb[j] + oub[j];
  else             v = cwb[j] + cub[j];
  out[t] = v;
}

// ---------------------------------------------------------------------------
// WMMA bf16 GEMM, double-buffered LDS staging.
//   Out(MxN) = epilogue( A(MxK) @ B(KxN) + bias ),  BT = B^T (N x K) bf16
//   256 threads = 8 waves; block tile 128x128; wave tile 64x32 (4x2 WMMA tiles)
// ---------------------------------------------------------------------------
__device__ __forceinline__ v16bf frag_a(const bf16_t* tile, int row, int hi) {
  // ISA 7.12.2 16-bit A 16x32: lane-half hi -> K runs [8hi, 8hi+8) and [16+8hi, +8)
  union { v16bf v; uint4 q[2]; } u;
  const bf16_t* p = tile + row * LDS_STRIDE;
  u.q[0] = *(const uint4*)(p + 8 * hi);
  u.q[1] = *(const uint4*)(p + 16 + 8 * hi);
  return u.v;
}
__device__ __forceinline__ v16bf frag_b(const bf16_t* tile, int row, int hi) {
  // B 32x16: lane-half hi -> K runs [8hi, 8hi+8) and [16+8hi, +8); tile holds B^T rows
  union { v16bf v; uint4 q[2]; } u;
  const bf16_t* p = tile + row * LDS_STRIDE;
  u.q[0] = *(const uint4*)(p + 8 * hi);
  u.q[1] = *(const uint4*)(p + 16 + 8 * hi);
  return u.v;
}

union Stage16 { uint4 q[2]; bf16_t e[16]; };

template<int A_F32, int RELU, int OUT_BF16>
__global__ __launch_bounds__(256) void gemm_wmma_bf16(
    const void* __restrict__ Aptr, const bf16_t* __restrict__ BT,
    const float* __restrict__ bias, void* __restrict__ Outp,
    int M, int N, int K, int lda)
{
  __shared__ __attribute__((aligned(16))) bf16_t As[2][128 * LDS_STRIDE];
  __shared__ __attribute__((aligned(16))) bf16_t Bs[2][128 * LDS_STRIDE];

  const int tid   = threadIdx.x;
  const int mBase = blockIdx.y * 128;
  const int nBase = blockIdx.x * 128;

  const int lr = tid >> 1;          // staging row 0..127
  const int lc = (tid & 1) * 16;    // staging col group: 0 or 16

  const int lane = tid & 31;
  const int wv   = tid >> 5;
  const int wm   = (wv >> 2) * 64;
  const int wn   = (wv & 3) * 32;
  const int l16  = lane & 15;
  const int hi   = lane >> 4;

  const float*  Af = (const float*)Aptr;
  const bf16_t* Ab = (const bf16_t*)Aptr;

  v8f acc[4][2];
  #pragma unroll
  for (int i = 0; i < 4; ++i)
    #pragma unroll
    for (int j = 0; j < 2; ++j)
      #pragma unroll
      for (int e = 0; e < 8; ++e) acc[i][j][e] = 0.f;

  // ---- chunk loaders: uniform fast path (k0+32<=K) is branch-free & vectorized
  auto loadA = [&](int k0, Stage16& s) {
    const int gk = k0 + lc;
    if (A_F32) {
      const float* ap = Af + (size_t)(mBase + lr) * lda + gk;
      if (k0 + 32 <= K) {
        #pragma unroll
        for (int i = 0; i < 16; ++i) s.e[i] = (bf16_t)ap[i];   // unconditional b32 loads
        if (k0 + 96 <= K) __builtin_prefetch(ap + 64, 0, 0);
      } else {
        #pragma unroll
        for (int i = 0; i < 16; ++i) s.e[i] = (bf16_t)((gk + i < K) ? ap[i] : 0.f);
      }
    } else {
      const bf16_t* ap = Ab + (size_t)(mBase + lr) * lda + gk;
      if (k0 + 32 <= K) {
        s.q[0] = *(const uint4*)(ap);
        s.q[1] = *(const uint4*)(ap + 8);
        if (k0 + 96 <= K) __builtin_prefetch(ap + 64, 0, 0);
      } else {
        #pragma unroll
        for (int i = 0; i < 16; ++i) s.e[i] = (gk + i < K) ? ap[i] : (bf16_t)0.f;
      }
    }
  };
  auto loadB = [&](int k0, Stage16& s) {
    const int gk = k0 + lc;
    const int gn = nBase + lr;
    if (gn < N) {
      const bf16_t* bp = BT + (size_t)gn * K + gk;
      if (k0 + 32 <= K) {
        s.q[0] = *(const uint4*)(bp);
        s.q[1] = *(const uint4*)(bp + 8);
        if (k0 + 96 <= K) __builtin_prefetch(bp + 64, 0, 0);
      } else {
        #pragma unroll
        for (int i = 0; i < 16; ++i) s.e[i] = (gk + i < K) ? bp[i] : (bf16_t)0.f;
      }
    } else {
      #pragma unroll
      for (int i = 0; i < 16; ++i) s.e[i] = (bf16_t)0.f;
    }
  };
  auto stage = [&](bf16_t* buf, const Stage16& s) {
    *(uint4*)(buf + lr * LDS_STRIDE + lc)     = s.q[0];
    *(uint4*)(buf + lr * LDS_STRIDE + lc + 8) = s.q[1];
  };

  const int nchunks = (K + 31) >> 5;
  Stage16 sa, sb;

  loadA(0, sa); loadB(0, sb);
  stage(As[0], sa); stage(Bs[0], sb);
  __syncthreads();

  for (int it = 0; it < nchunks; ++it) {
    const int cur = it & 1;
    const bool more = (it + 1 < nchunks);
    if (more) { loadA((it + 1) << 5, sa); loadB((it + 1) << 5, sb); }  // overlap with WMMA

    v16bf a[4], b[2];
    #pragma unroll
    for (int i = 0; i < 4; ++i) a[i] = frag_a(As[cur], wm + i * 16 + l16, hi);
    #pragma unroll
    for (int j = 0; j < 2; ++j) b[j] = frag_b(Bs[cur], wn + j * 16 + l16, hi);
    #pragma unroll
    for (int i = 0; i < 4; ++i)
      #pragma unroll
      for (int j = 0; j < 2; ++j)
        acc[i][j] = __builtin_amdgcn_wmma_f32_16x16x32_bf16(
            false, a[i], false, b[j], (short)0, acc[i][j], false, false);

    if (more) { stage(As[cur ^ 1], sa); stage(Bs[cur ^ 1], sb); }
    __syncthreads();
  }

  // ---- epilogue: bias (+ReLU); C/D layout: VGPR r -> M = r + 8*hi, N = l16
  float*  Of = (float*)Outp;
  bf16_t* Ob = (bf16_t*)Outp;
  #pragma unroll
  for (int i = 0; i < 4; ++i) {
    #pragma unroll
    for (int j = 0; j < 2; ++j) {
      const int gcol = nBase + wn + j * 16 + l16;
      if (gcol < N) {
        const float bv = bias[gcol];
        const int rbase = mBase + wm + i * 16 + hi * 8;
        #pragma unroll
        for (int r = 0; r < 8; ++r) {
          float v = acc[i][j][r] + bv;
          if (RELU) v = v > 0.f ? v : 0.f;
          const size_t o = (size_t)(rbase + r) * N + gcol;
          if (OUT_BF16) Ob[o] = (bf16_t)v; else Of[o] = v;
        }
      }
    }
  }
}

// ---------------------------------------------------------------------------
// Sequential recurrence: one persistent 768-thread workgroup, h in LDS.
// ew: (T x 768) fp32 = e@W + (w_b+u_b), gate-major cols [i|o|c]
// UcatT: (768 x 256) bf16, row (g*256+j) = column j of U_g
// ---------------------------------------------------------------------------
__global__ __launch_bounds__(768) void recurrence_kernel(
    const float* __restrict__ ew, const bf16_t* __restrict__ UcatT,
    bf16_t* __restrict__ hsbf, int T)
{
  __shared__ float h[256];
  __shared__ float pre[768];
  const int t = threadIdx.x;
  if (t < 256) h[t] = 0.f;
  __syncthreads();

  const bf16_t* urow = UcatT + (size_t)t * 256;

  for (int s = 0; s < T; ++s) {
    float acc = 0.f;
    #pragma unroll 4
    for (int k = 0; k < 256; k += 8) {
      union { uint4 q; bf16_t e[8]; } ub;
      ub.q = *(const uint4*)(urow + k);      // 16B/lane, L2-resident (384KB total)
      #pragma unroll
      for (int i = 0; i < 8; ++i) acc += (float)ub.e[i] * h[k + i];  // LDS broadcast
    }
    pre[t] = acc + ew[(size_t)s * 768 + t];
    __syncthreads();
    if (t < 256) {
      const float iv = 1.f / (1.f + __expf(-pre[t]));
      const float ov = 1.f / (1.f + __expf(-pre[256 + t]));
      const float ct = tanhf(pre[512 + t]);
      const float hn = ov * tanhf(iv * ct);   // c is recomputed, not carried
      h[t] = hn;
      hsbf[(size_t)s * 256 + t] = (bf16_t)hn;
    }
    __syncthreads();
  }
}

// ---------------------------------------------------------------------------
// Host-side orchestration
// ---------------------------------------------------------------------------
static inline size_t align256(size_t x) { return (x + 255) & ~(size_t)255; }

extern "C" void kernel_launch(void* const* d_in, const int* in_sizes, int n_in,
                              void* d_out, int out_size, void* d_ws, size_t ws_size,
                              hipStream_t stream) {
  (void)in_sizes; (void)n_in; (void)out_size;
  const int B = 4096, N = 10000, H = 512, E = 256;

  const float* x      = (const float*)d_in[0];
  const float* enc_w1 = (const float*)d_in[1];
  const float* enc_b1 = (const float*)d_in[2];
  const float* enc_w2 = (const float*)d_in[3];
  const float* enc_b2 = (const float*)d_in[4];
  const float* i_u_w  = (const float*)d_in[5];
  const float* i_u_b  = (const float*)d_in[6];
  const float* i_w_w  = (const float*)d_in[7];
  const float* i_w_b  = (const float*)d_in[8];
  const float* o_u_w  = (const float*)d_in[9];
  const float* o_u_b  = (const float*)d_in[10];
  const float* o_w_w  = (const float*)d_in[11];
  const float* o_w_b  = (const float*)d_in[12];
  const float* c_u_w  = (const float*)d_in[13];
  const float* c_u_b  = (const float*)d_in[14];
  const float* c_w_w  = (const float*)d_in[15];
  const float* c_w_b  = (const float*)d_in[16];
  const float* dec_w1 = (const float*)d_in[17];
  const float* dec_b1 = (const float*)d_in[18];
  const float* dec_w2 = (const float*)d_in[19];
  const float* dec_b2 = (const float*)d_in[20];

  char* ws = (char*)d_ws;
  size_t off = 0;
  auto take = [&](size_t bytes) -> void* { void* p = ws + off; off += align256(bytes); return p; };

  bf16_t* w1T   = (bf16_t*)take((size_t)H * N * 2);
  bf16_t* t1    = (bf16_t*)take((size_t)B * H * 2);
  bf16_t* w2T   = (bf16_t*)take((size_t)E * H * 2);
  bf16_t* embbf = (bf16_t*)take((size_t)B * E * 2);
  bf16_t* WcatT = (bf16_t*)take((size_t)3 * E * E * 2);
  float*  gbias = (float*) take((size_t)3 * E * 4);
  float*  ew    = (float*) take((size_t)B * 3 * E * 4);
  bf16_t* UcatT = (bf16_t*)take((size_t)3 * E * E * 2);
  bf16_t* hsbf  = (bf16_t*)take((size_t)B * E * 2);
  bf16_t* d1T   = (bf16_t*)take((size_t)H * E * 2);
  bf16_t* t2    = (bf16_t*)take((size_t)B * H * 2);
  bf16_t* w2dT  = (bf16_t*)take((size_t)N * H * 2);
  const size_t xbf_bytes = (size_t)B * N * 2;            // 82 MB, allocated last
  bf16_t* xbf   = (bf16_t*)(ws + off);
  const bool use_xbf = (ws_size >= off + xbf_bytes);     // deterministic per run

  auto t2d = [](int K, int N_) { return dim3((K + 31) / 32, (N_ + 31) / 32); };

  // --- one-time weight transposes to (N x K) bf16
  transpose_to_bf16<<<t2d(N, H), 256, 0, stream>>>(enc_w1, w1T, N, H);
  transpose_to_bf16<<<t2d(H, E), 256, 0, stream>>>(enc_w2, w2T, H, E);
  transpose_to_bf16<<<t2d(E, E), 256, 0, stream>>>(i_w_w, WcatT + 0 * E * E, E, E);
  transpose_to_bf16<<<t2d(E, E), 256, 0, stream>>>(o_w_w, WcatT + 1 * E * E, E, E);
  transpose_to_bf16<<<t2d(E, E), 256, 0, stream>>>(c_w_w, WcatT + 2 * E * E, E, E);
  transpose_to_bf16<<<t2d(E, E), 256, 0, stream>>>(i_u_w, UcatT + 0 * E * E, E, E);
  transpose_to_bf16<<<t2d(E, E), 256, 0, stream>>>(o_u_w, UcatT + 1 * E * E, E, E);
  transpose_to_bf16<<<t2d(E, E), 256, 0, stream>>>(c_u_w, UcatT + 2 * E * E, E, E);
  transpose_to_bf16<<<t2d(E, H), 256, 0, stream>>>(dec_w1, d1T, E, H);
  transpose_to_bf16<<<t2d(H, N), 256, 0, stream>>>(dec_w2, w2dT, H, N);
  gate_bias_kernel<<<1, 768, 0, stream>>>(i_w_b, i_u_b, o_w_b, o_u_b, c_w_b, c_u_b, gbias);

  // --- encoder GEMM1: t1 = relu(x[:, :N] @ enc_w1 + b1)
  if (use_xbf) {
    convert_x_bf16<<<(B * N + 255) / 256, 256, 0, stream>>>(x, xbf, B, N, N + 1);
    gemm_wmma_bf16<0, 1, 1><<<dim3(H / 128, B / 128), 256, 0, stream>>>(
        xbf, w1T, enc_b1, t1, B, H, N, N);
  } else {
    gemm_wmma_bf16<1, 1, 1><<<dim3(H / 128, B / 128), 256, 0, stream>>>(
        x, w1T, enc_b1, t1, B, H, N, N + 1);
  }
  // --- emb = t1 @ enc_w2 + b2 (bf16 out)
  gemm_wmma_bf16<0, 0, 1><<<dim3(E / 128, B / 128), 256, 0, stream>>>(
      t1, w2T, enc_b2, embbf, B, E, H, H);
  // --- hoisted gate input projections: ew = emb @ [Wi|Wo|Wc] + (w_b+u_b)
  gemm_wmma_bf16<0, 0, 0><<<dim3(3 * E / 128, B / 128), 256, 0, stream>>>(
      embbf, WcatT, gbias, ew, B, 3 * E, E, E);
  // --- sequential scan (latency-bound; only 3 h@U matvecs per step remain)
  recurrence_kernel<<<1, 768, 0, stream>>>(ew, UcatT, hsbf, B);
  // --- decoder: t2 = relu(hs @ dec_w1 + b1)
  gemm_wmma_bf16<0, 1, 1><<<dim3(H / 128, B / 128), 256, 0, stream>>>(
      hsbf, d1T, dec_b1, t2, B, H, E, E);
  // --- out = t2 @ dec_w2 + b2 (fp32, N=10000 edge-guarded)
  gemm_wmma_bf16<0, 0, 0><<<dim3((N + 127) / 128, B / 128), 256, 0, stream>>>(
      t2, w2dT, dec_b2, d_out, B, N, H, H);
}
